// TFCLIPAttention_79620103733759
// MI455X (gfx1250) — matrix operand order repaired and checked by reading, not compiled, not run on hardware
//
#include <hip/hip_runtime.h>
#include <hip/hip_bf16.h>

// ---------- types ----------
typedef __attribute__((ext_vector_type(16))) __bf16         v16bf;
typedef __attribute__((ext_vector_type(8)))  float          v8f;
typedef __attribute__((ext_vector_type(4)))  unsigned int   uv4;
typedef __attribute__((ext_vector_type(4)))  float          fv4;
typedef int v4i_vs __attribute__((vector_size(16)));        // matches builtin prototype

struct Frag32B { uv4 lo, hi; };

#define BATCH 4
#define SEQ   2048
#define DMODEL 768
#define NHEAD 12
#define HDIM  64
#define MROWS (BATCH * SEQ)   // 8192

// ---------- gfx1250 async global->LDS copies (ASYNCcnt) ----------
#if defined(__gfx1250__) && __has_builtin(__builtin_amdgcn_global_load_async_to_lds_b128)
#define HAVE_ASYNC_LDS 1
#else
#define HAVE_ASYNC_LDS 0
#endif

__device__ __forceinline__ void copy16B_g2lds(const unsigned short* gsrc, unsigned short* ldst) {
#if HAVE_ASYNC_LDS
    __builtin_amdgcn_global_load_async_to_lds_b128(
        (__attribute__((address_space(1))) v4i_vs*)gsrc,
        (__attribute__((address_space(3))) v4i_vs*)ldst, 0, 0);
#else
    *(uv4*)ldst = *(const uv4*)gsrc;
#endif
}

__device__ __forceinline__ void wait_async_lds() {
#if HAVE_ASYNC_LDS
#if __has_builtin(__builtin_amdgcn_s_wait_asynccnt)
    __builtin_amdgcn_s_wait_asynccnt(0);
#else
    asm volatile("s_wait_asynccnt 0" ::: "memory");
#endif
#endif
}

__device__ __forceinline__ unsigned short f2bf(float f) {
    union { float f; unsigned u; } v; v.f = f;
    unsigned r = v.u + 0x7FFFu + ((v.u >> 16) & 1u);
    return (unsigned short)(r >> 16);
}

__device__ __forceinline__ v8f wmma_bf16(v16bf a, v16bf b, v8f c) {
    return __builtin_amdgcn_wmma_f32_16x16x32_bf16(false, a, false, b, (short)0, c, false, false);
}

// A fragment (16x32, bf16): lane holds row m0+(lane&15); elements 0..7 -> k=k0+g*8+e,
// elements 8..15 -> k=k0+16+g*8+(e-8), g=lane>>4.  Two 16B LDS loads.
__device__ __forceinline__ v16bf load_a_frag(const unsigned short* tile, int lda, int m0, int k0) {
    int lane = threadIdx.x & 31;
    const unsigned short* row = tile + (m0 + (lane & 15)) * lda + k0 + ((lane >> 4) << 3);
    Frag32B t;
    t.lo = *(const uv4*)(row);
    t.hi = *(const uv4*)(row + 16);
    return __builtin_bit_cast(v16bf, t);
}

// B fragment (32x16, bf16) read from a TRANSPOSED tile Bt[n][k]: lane holds col n0+(lane&15),
// elements 0..7 -> k=k0+g*8+e, elements 8..15 -> k=k0+16+g*8+(e-8), g=lane>>4.  Two 16B LDS loads.
__device__ __forceinline__ v16bf load_b_frag(const unsigned short* tile, int ldb, int n0, int k0) {
    int lane = threadIdx.x & 31;
    const unsigned short* row = tile + (n0 + (lane & 15)) * ldb + k0 + ((lane >> 4) << 3);
    Frag32B t;
    t.lo = *(const uv4*)(row);
    t.hi = *(const uv4*)(row + 16);
    return __builtin_bit_cast(v16bf, t);
}

// =====================================================================
// Kernel 1: fused QKV projection.  X[8192x768]f32 @ W[768x768]f32 + b.
// grid = (64, 36): x -> 128-row tile, y -> (sel 0..2)*12 + head.
// Q,K stored bf16 [bh][s][hd]; V stored bf16 transposed [bh][hd][s].
// Q pre-scaled by 1/sqrt(64).
// =====================================================================
__global__ __launch_bounds__(256) void qkv_kernel(
    const float* __restrict__ X,
    const float* __restrict__ Wq, const float* __restrict__ bq,
    const float* __restrict__ Wk, const float* __restrict__ bk,
    const float* __restrict__ Wv, const float* __restrict__ bv,
    unsigned short* __restrict__ Qo, unsigned short* __restrict__ Ko,
    unsigned short* __restrict__ Vo)
{
    constexpr int LDA = 40;   // 32 + 8 pad (halfs)
    constexpr int LDB = 40;
    __shared__ __align__(16) unsigned short As[128 * LDA];
    __shared__ __align__(16) unsigned short Bt[64 * LDB];   // transposed W tile [n][k]

    const int tid  = threadIdx.x;
    const int wave = tid >> 5;
    const int lane = tid & 31;
    const int m0   = blockIdx.x * 128;
    const int sel  = blockIdx.y / NHEAD;       // 0=Q 1=K 2=V
    const int head = blockIdx.y % NHEAD;
    const int n0   = head * HDIM;

    const float* W    = (sel == 0) ? Wq : (sel == 1) ? Wk : Wv;
    const float* bias = (sel == 0) ? bq : (sel == 1) ? bk : bv;

    v8f acc[4];
#pragma unroll
    for (int i = 0; i < 4; ++i) acc[i] = (v8f)0.0f;

    for (int k0 = 0; k0 < DMODEL; k0 += 32) {
        // A tile: 128x32 f32 -> bf16 (1024 float4 loads, 4 per thread)
#pragma unroll
        for (int j = 0; j < 4; ++j) {
            int e   = tid + j * 256;
            int row = e >> 3;
            int c4  = (e & 7) << 2;
            fv4 x = *(const fv4*)(X + (size_t)(m0 + row) * DMODEL + k0 + c4);
            unsigned lo = (unsigned)f2bf(x[0]) | ((unsigned)f2bf(x[1]) << 16);
            unsigned hi = (unsigned)f2bf(x[2]) | ((unsigned)f2bf(x[3]) << 16);
            unsigned* dst = (unsigned*)(As + row * LDA + c4);
            dst[0] = lo; dst[1] = hi;
        }
        // B tile: W[k0..k0+32)[n0..n0+64) -> transposed Bt[n][k]
#pragma unroll
        for (int j = 0; j < 2; ++j) {
            int e  = tid + j * 256;
            int kr = e >> 4;
            int c4 = (e & 15) << 2;
            fv4 w = *(const fv4*)(W + (size_t)(k0 + kr) * DMODEL + n0 + c4);
            Bt[(c4 + 0) * LDB + kr] = f2bf(w[0]);
            Bt[(c4 + 1) * LDB + kr] = f2bf(w[1]);
            Bt[(c4 + 2) * LDB + kr] = f2bf(w[2]);
            Bt[(c4 + 3) * LDB + kr] = f2bf(w[3]);
        }
        __syncthreads();

        v16bf a = load_a_frag(As, LDA, wave * 16, 0);
        v16bf bfr[4];
#pragma unroll
        for (int nt = 0; nt < 4; ++nt) bfr[nt] = load_b_frag(Bt, LDB, nt * 16, 0);
#pragma unroll
        for (int nt = 0; nt < 4; ++nt) acc[nt] = wmma_bf16(a, bfr[nt], acc[nt]);
        __syncthreads();
    }

    // epilogue: bias, optional 1/8 scale, scatter to Q/K/V layouts
    const float qscale = 0.125f;   // 1/sqrt(64)
    if (sel == 0) {
#pragma unroll
        for (int nt = 0; nt < 4; ++nt)
#pragma unroll
            for (int r = 0; r < 8; ++r) {
                int grow = m0 + wave * 16 + r + ((lane >> 4) << 3);
                int col  = n0 + nt * 16 + (lane & 15);
                float v  = (acc[nt][r] + bias[col]) * qscale;
                int b = grow / SEQ, s = grow % SEQ;
                Qo[((size_t)(b * NHEAD + head) * SEQ + s) * HDIM + (col - n0)] = f2bf(v);
            }
    } else if (sel == 1) {
#pragma unroll
        for (int nt = 0; nt < 4; ++nt)
#pragma unroll
            for (int r = 0; r < 8; ++r) {
                int grow = m0 + wave * 16 + r + ((lane >> 4) << 3);
                int col  = n0 + nt * 16 + (lane & 15);
                float v  = acc[nt][r] + bias[col];
                int b = grow / SEQ, s = grow % SEQ;
                Ko[((size_t)(b * NHEAD + head) * SEQ + s) * HDIM + (col - n0)] = f2bf(v);
            }
    } else {
#pragma unroll
        for (int nt = 0; nt < 4; ++nt)
#pragma unroll
            for (int r = 0; r < 8; ++r) {
                int grow = m0 + wave * 16 + r + ((lane >> 4) << 3);
                int col  = n0 + nt * 16 + (lane & 15);
                float v  = acc[nt][r] + bias[col];
                int b = grow / SEQ, s = grow % SEQ;
                Vo[((size_t)(b * NHEAD + head) * HDIM + (col - n0)) * SEQ + s] = f2bf(v);
            }
    }
}

// =====================================================================
// Kernel 2: flash-style attention.  grid = (32 qblocks, 48 bh), 128 thr.
// Each wave owns 16 query rows; online softmax in fp32; P,V via WMMA.
// ctx written bf16 to CTX[b*2048+q][h*64+hd]  (i.e. [8192 x 768]).
// =====================================================================
__global__ __launch_bounds__(128) void attn_kernel(
    const unsigned short* __restrict__ Q,
    const unsigned short* __restrict__ K,
    const unsigned short* __restrict__ V,
    unsigned short* __restrict__ CTX)
{
    constexpr int LD = 72;   // 64 + 8 pad (halfs)
    __shared__ __align__(16) unsigned short Qs[64 * LD];
    __shared__ __align__(16) unsigned short Ks[64 * LD];
    __shared__ __align__(16) unsigned short Vt[64 * LD];   // [hd][key]
    __shared__ __align__(16) unsigned short Ps[64 * LD];   // [q][key]

    const int tid  = threadIdx.x;
    const int wave = tid >> 5;
    const int lane = tid & 31;
    const int qb   = blockIdx.x;      // query block (64 rows)
    const int bh   = blockIdx.y;      // b*12+h

    // load Q tile [64 q][64 hd] (bf16, contiguous) via async global->LDS
    {
        const unsigned short* src = Q + ((size_t)bh * SEQ + qb * 64) * HDIM;
#pragma unroll
        for (int j = 0; j < 4; ++j) {
            int e = tid + j * 128;
            int row = e >> 3;
            int c8  = (e & 7) << 3;
            copy16B_g2lds(src + row * HDIM + c8, Qs + row * LD + c8);
        }
    }

    float mx[8], sm[8];
    v8f cacc[4];
#pragma unroll
    for (int r = 0; r < 8; ++r) { mx[r] = -1e30f; sm[r] = 0.0f; }
#pragma unroll
    for (int i = 0; i < 4; ++i) cacc[i] = (v8f)0.0f;

    for (int kb = 0; kb < SEQ / 64; ++kb) {
        // K tile [key][hd] natural; V tile [hd][key] from transposed V
        {
            const unsigned short* ksrc = K + ((size_t)bh * SEQ + kb * 64) * HDIM;
#pragma unroll
            for (int j = 0; j < 4; ++j) {
                int e = tid + j * 128;
                int row = e >> 3;
                int c8  = (e & 7) << 3;
                copy16B_g2lds(ksrc + row * HDIM + c8, Ks + row * LD + c8);
            }
            const unsigned short* vsrc = V + (size_t)bh * HDIM * SEQ + kb * 64;
#pragma unroll
            for (int j = 0; j < 4; ++j) {
                int e = tid + j * 128;
                int row = e >> 3;          // hd
                int c8  = (e & 7) << 3;    // key offset
                copy16B_g2lds(vsrc + (size_t)row * SEQ + c8, Vt + row * LD + c8);
            }
        }
        wait_async_lds();
        __syncthreads();

        // scores: S[16q x 64key] = Q . K^T   (contraction over hd, 2 chunks of 32)
        v8f sc4[4];
        {
            v16bf a0 = load_a_frag(Qs, LD, wave * 16, 0);
            v16bf a1 = load_a_frag(Qs, LD, wave * 16, 32);
            v16bf b0[4], b1[4];
#pragma unroll
            for (int nt = 0; nt < 4; ++nt) {
                b0[nt] = load_b_frag(Ks, LD, nt * 16, 0);
                b1[nt] = load_b_frag(Ks, LD, nt * 16, 32);
            }
#pragma unroll
            for (int nt = 0; nt < 4; ++nt) {
                v8f s = (v8f)0.0f;
                s = wmma_bf16(a0, b0[nt], s);
                s = wmma_bf16(a1, b1[nt], s);
                sc4[nt] = s;
            }
        }

        // online softmax per row r (row = wave*16 + r + 8*(lane>>4))
#pragma unroll
        for (int r = 0; r < 8; ++r) {
            float rv = sc4[0][r];
            rv = fmaxf(rv, sc4[1][r]);
            rv = fmaxf(rv, sc4[2][r]);
            rv = fmaxf(rv, sc4[3][r]);
#pragma unroll
            for (int off = 1; off < 16; off <<= 1)
                rv = fmaxf(rv, __shfl_xor(rv, off, 32));
            float newm = fmaxf(mx[r], rv);
            float scale = __expf(mx[r] - newm);

            int prow = wave * 16 + r + ((lane >> 4) << 3);
            float psum = 0.0f;
#pragma unroll
            for (int nt = 0; nt < 4; ++nt) {
                float p = __expf(sc4[nt][r] - newm);
                psum += p;
                Ps[prow * LD + nt * 16 + (lane & 15)] = f2bf(p);
            }
#pragma unroll
            for (int off = 1; off < 16; off <<= 1)
                psum += __shfl_xor(psum, off, 32);

            sm[r] = sm[r] * scale + psum;
            mx[r] = newm;
#pragma unroll
            for (int nt = 0; nt < 4; ++nt) cacc[nt][r] *= scale;
        }

        // ctx += P @ V : contraction over keys (64, 2 chunks of 32)
        {
            v16bf a0 = load_a_frag(Ps, LD, wave * 16, 0);
            v16bf a1 = load_a_frag(Ps, LD, wave * 16, 32);
            v16bf b0[4], b1[4];
#pragma unroll
            for (int nt = 0; nt < 4; ++nt) {
                b0[nt] = load_b_frag(Vt, LD, nt * 16, 0);
                b1[nt] = load_b_frag(Vt, LD, nt * 16, 32);
            }
#pragma unroll
            for (int nt = 0; nt < 4; ++nt) {
                cacc[nt] = wmma_bf16(a0, b0[nt], cacc[nt]);
                cacc[nt] = wmma_bf16(a1, b1[nt], cacc[nt]);
            }
        }
        __syncthreads();
    }

    // normalize and write ctx (bf16) to [b*2048+q][h*64+hd]
    const int h = bh % NHEAD;
    const int b = bh / NHEAD;
#pragma unroll
    for (int r = 0; r < 8; ++r) {
        float inv = 1.0f / sm[r];
        int q = qb * 64 + wave * 16 + r + ((lane >> 4) << 3);
#pragma unroll
        for (int nt = 0; nt < 4; ++nt) {
            int hd = nt * 16 + (lane & 15);
            CTX[((size_t)(b * SEQ + q)) * DMODEL + h * HDIM + hd] = f2bf(cacc[nt][r] * inv);
        }
    }
}

// =====================================================================
// Kernel 3: output projection.  CTX[8192x768]bf16 @ Wo[768x768]f32 + bo
// -> out f32.  grid = (64, 12), 256 threads.
// =====================================================================
__global__ __launch_bounds__(256) void oproj_kernel(
    const unsigned short* __restrict__ A,
    const float* __restrict__ Wo, const float* __restrict__ bo,
    float* __restrict__ out)
{
    constexpr int LDA = 40;
    constexpr int LDB = 40;
    __shared__ __align__(16) unsigned short As[128 * LDA];
    __shared__ __align__(16) unsigned short Bt[64 * LDB];

    const int tid  = threadIdx.x;
    const int wave = tid >> 5;
    const int lane = tid & 31;
    const int m0   = blockIdx.x * 128;
    const int n0   = blockIdx.y * 64;

    v8f acc[4];
#pragma unroll
    for (int i = 0; i < 4; ++i) acc[i] = (v8f)0.0f;

    for (int k0 = 0; k0 < DMODEL; k0 += 32) {
        // A tile 128x32 bf16: async direct copies (512 x 16B, 2 per thread)
#pragma unroll
        for (int j = 0; j < 2; ++j) {
            int e   = tid + j * 256;
            int row = e >> 2;
            int c8  = (e & 3) << 3;
            copy16B_g2lds(A + (size_t)(m0 + row) * DMODEL + k0 + c8, As + row * LDA + c8);
        }
        // B tile 32x64 f32 -> bf16 transposed
#pragma unroll
        for (int j = 0; j < 2; ++j) {
            int e  = tid + j * 256;
            int kr = e >> 4;
            int c4 = (e & 15) << 2;
            fv4 w = *(const fv4*)(Wo + (size_t)(k0 + kr) * DMODEL + n0 + c4);
            Bt[(c4 + 0) * LDB + kr] = f2bf(w[0]);
            Bt[(c4 + 1) * LDB + kr] = f2bf(w[1]);
            Bt[(c4 + 2) * LDB + kr] = f2bf(w[2]);
            Bt[(c4 + 3) * LDB + kr] = f2bf(w[3]);
        }
        wait_async_lds();
        __syncthreads();

        v16bf a = load_a_frag(As, LDA, wave * 16, 0);
        v16bf bfr[4];
#pragma unroll
        for (int nt = 0; nt < 4; ++nt) bfr[nt] = load_b_frag(Bt, LDB, nt * 16, 0);
#pragma unroll
        for (int nt = 0; nt < 4; ++nt) acc[nt] = wmma_bf16(a, bfr[nt], acc[nt]);
        __syncthreads();
    }

#pragma unroll
    for (int nt = 0; nt < 4; ++nt)
#pragma unroll
        for (int r = 0; r < 8; ++r) {
            int grow = m0 + wave * 16 + r + ((lane >> 4) << 3);
            int col  = n0 + nt * 16 + (lane & 15);
            out[(size_t)grow * DMODEL + col] = acc[nt][r] + bo[col];
        }
}

// =====================================================================
extern "C" void kernel_launch(void* const* d_in, const int* in_sizes, int n_in,
                              void* d_out, int out_size, void* d_ws, size_t ws_size,
                              hipStream_t stream) {
    const float* X  = (const float*)d_in[0];
    const float* Wq = (const float*)d_in[1];
    const float* bq = (const float*)d_in[2];
    const float* Wk = (const float*)d_in[3];
    const float* bk = (const float*)d_in[4];
    const float* Wv = (const float*)d_in[5];
    const float* bv = (const float*)d_in[6];
    const float* Wo = (const float*)d_in[7];
    const float* bo = (const float*)d_in[8];
    float* out = (float*)d_out;

    const size_t qElems = (size_t)BATCH * NHEAD * SEQ * HDIM;   // 6,291,456
    unsigned short* Qw  = (unsigned short*)d_ws;
    unsigned short* Kw  = Qw + qElems;
    unsigned short* Vw  = Kw + qElems;
    unsigned short* Cw  = Vw + qElems;

    qkv_kernel<<<dim3(MROWS / 128, 3 * NHEAD), 256, 0, stream>>>(
        X, Wq, bq, Wk, bk, Wv, bv, Qw, Kw, Vw);
    attn_kernel<<<dim3(SEQ / 64, BATCH * NHEAD), 128, 0, stream>>>(Qw, Kw, Vw, Cw);
    oproj_kernel<<<dim3(MROWS / 128, DMODEL / 64), 256, 0, stream>>>(Cw, Wo, bo, out);
}
